// ClientModel_61692910240099
// MI455X (gfx1250) — matrix operand order repaired and checked by reading, not compiled, not run on hardware
//
#include <hip/hip_runtime.h>
#include <hip/hip_bf16.h>

#define SEQ_LEN 80
#define NUM_CLASSES 80
#define HID 256
#define EMB 8
#define BATCH 2048

typedef __attribute__((ext_vector_type(16))) __bf16 v16bf;
typedef __attribute__((ext_vector_type(8)))  float  v8f;

// ---------- helpers ----------

__device__ __forceinline__ unsigned short f2bf(float f) {
    unsigned u = __float_as_uint(f);
    u += 0x7fffu + ((u >> 16) & 1u);   // round-to-nearest-even
    return (unsigned short)(u >> 16);
}

__device__ __forceinline__ float sigm(float x) {
    return 1.0f / (1.0f + __expf(-x));
}

__device__ __forceinline__ float tanh_fast(float x) {
    float e = __expf(2.0f * x);
    return 1.0f - 2.0f / (e + 1.0f);
}

// Store scalar into a 16-bit WMMA A-fragment (16x32 tile) held as ushort[512]
// per k-tile. Mapping per CDNA5 ISA 7.12.2 (16-bit A-matrix 16x32):
//   lanes 0-15: V0-3 hold K=0..7, V4-7 hold K=16..23
//   lanes16-31: V0-3 hold K=8..15, V4-7 hold K=24..31
__device__ __forceinline__ void st_afrag(unsigned short* A, int k, int m,
                                         unsigned short v) {
    int kt  = k >> 5;
    int kk  = k & 31;
    int hi  = kk >> 4;          // upper 16 of the K-tile -> VGPRs 4..7
    int w   = kk & 15;
    int lh  = (w >> 3) & 1;     // lane half
    int w8  = w & 7;
    int e   = 2 * (4 * hi + (w8 >> 1)) + (kk & 1);
    int ln  = m + 16 * lh;
    A[kt * 512 + ln * 16 + e] = v;
}

// ---------- weight pre-conversion (f32 row-major -> bf16 B-fragments) ----------
// B-fragment (32x16, 16-bit) layout: lanes 0-15 hold K=0..7 (elem 0-7), K=16..23 (elem 8-15),
// lanes 16-31 hold K=8..15 (elem 0-7), K=24..31 (elem 8-15), column N = lane%16.

__global__ void conv_w_kernel(const float* __restrict__ W, unsigned short* __restrict__ Wf,
                              int ktiles, int ntiles, int ncols, int krows) {
    int idx = blockIdx.x * blockDim.x + threadIdx.x;
    int total = ktiles * ntiles * 512;
    if (idx >= total) return;
    int le   = idx & 511;
    int rem  = idx >> 9;
    int nt   = rem % ntiles;
    int kt   = rem / ntiles;
    int lane = le >> 4;
    int e    = le & 15;
    int kk   = ((e >> 3) << 4) + ((lane >> 4) << 3) + (e & 7);
    int k    = kt * 32 + kk;
    int n    = nt * 16 + (lane & 15);
    float v  = (k < krows) ? W[k * ncols + n] : 0.0f;
    Wf[idx] = f2bf(v);
}

// ---------- main persistent sequence kernel ----------

__global__ __launch_bounds__(256)
void lstm_seq_kernel(const int*   __restrict__ feats,
                     const float* __restrict__ emb,
                     const unsigned short* __restrict__ W1f,  // [9][64][512]
                     const float* __restrict__ b1,
                     const unsigned short* __restrict__ W2f,  // [16][64][512]
                     const float* __restrict__ b2,
                     const unsigned short* __restrict__ Wdf,  // [8][5][512]
                     const float* __restrict__ bd,
                     float* __restrict__ out) {
    // A1: layer1 A fragments, K = [x(8) | h1(256) | pad(24)] -> 9 k-tiles
    // A2: layer2 A fragments, K = [h1(256) | h2(256)]        -> 16 k-tiles
    __shared__ __attribute__((aligned(32))) unsigned short A1[9 * 512];
    __shared__ __attribute__((aligned(32))) unsigned short A2[16 * 512];

    const int tid  = threadIdx.x;
    const int lane = tid & 31;
    const int wv   = tid >> 5;      // wave 0..7
    const int bid  = blockIdx.x;    // batch tile: rows [bid*16, bid*16+16)
    const int lnN  = lane & 15;
    const int lh   = lane >> 4;

    for (int i = tid; i < 9 * 512; i += 256)  A1[i] = 0;
    for (int i = tid; i < 16 * 512; i += 256) A2[i] = 0;

    float c1[2][8], c2[2][8];
#pragma unroll
    for (int b = 0; b < 2; ++b)
#pragma unroll
        for (int r = 0; r < 8; ++r) { c1[b][r] = 0.0f; c2[b][r] = 0.0f; }

    // wave wv owns N-tiles {wv, wv+8, ..., wv+56}; bias per tile is per-lane
    float bias1[8], bias2[8];
#pragma unroll
    for (int q = 0; q < 8; ++q) {
        int col = (wv + 8 * q) * 16 + lnN;
        bias1[q] = b1[col];
        bias2[q] = b2[col];
    }
    __syncthreads();

    // Opaque SGPR element-offset (always 0): redefined by the asm each
    // iteration so the (loop-invariant) weight loads cannot be hoisted out of
    // the t-loop and spilled to scratch, while the weight base pointers keep
    // their global-address-space provenance -> global_load_b128 saddr.
    unsigned woff = 0;

    v8f acc[8];

    for (int t = 0; t < SEQ_LEN; ++t) {
        asm volatile("" : "+s"(woff));

        // ---- stage x_t (embedding lookup) into A1 k-tile 0, K=0..7 ----
        // for k<8: A-frag position is lane=m, elem=k
        if (tid < 128) {
            int row = tid >> 3, kx = tid & 7;
            int f = feats[(bid * 16 + row) * SEQ_LEN + t];
            A1[row * 16 + kx] = f2bf(emb[f * EMB + kx]);
        }
        __syncthreads();

        // ---- layer 1 GEMM: z1 = [x,h1] @ W1  (16 x 288 x 1024) ----
#pragma unroll
        for (int q = 0; q < 8; ++q) {
#pragma unroll
            for (int r = 0; r < 8; ++r) acc[q][r] = 0.0f;
        }
        for (int kt = 0; kt < 9; ++kt) {
            v16bf a = *(const v16bf*)&A1[kt * 512 + lane * 16];
            v16bf bfr[8];
#pragma unroll
            for (int q = 0; q < 8; ++q)
                bfr[q] = *(const v16bf*)&W1f[woff + (kt * 64 + (wv + 8 * q)) * 512 + lane * 16];
            // Issue the whole load batch before any WMMA: one L2 round-trip
            // per k-tile, WMMAs drain with partial loadcnt waits.
            __builtin_amdgcn_sched_barrier(0);
#pragma unroll
            for (int q = 0; q < 8; ++q)
                acc[q] = __builtin_amdgcn_wmma_f32_16x16x32_bf16(
                    false, a, false, bfr[q], (short)0, acc[q], false, false);
        }
        __syncthreads();   // all reads of A1 done before h1 rewrite

        // ---- layer 1 cell update; write h1 (bf16) into A1 and A2 ----
#pragma unroll
        for (int blk = 0; blk < 2; ++blk) {
            int n = (wv + 8 * blk) * 16 + lnN;   // h-column, 0..255
#pragma unroll
            for (int r = 0; r < 8; ++r) {
                float iv = acc[blk][r]     + bias1[blk];
                float jv = acc[blk + 2][r] + bias1[blk + 2];
                float fv = acc[blk + 4][r] + bias1[blk + 4];
                float ov = acc[blk + 6][r] + bias1[blk + 6];
                float nc = c1[blk][r] * sigm(fv + 1.0f) + sigm(iv) * tanh_fast(jv);
                float nh = tanh_fast(nc) * sigm(ov);
                c1[blk][r] = nc;
                unsigned short hb = f2bf(nh);
                int m = r + 8 * lh;
                st_afrag(A2, n,     m, hb);  // layer2 input, K = n
                st_afrag(A1, 8 + n, m, hb);  // next-step layer1 input
            }
        }
        __syncthreads();

        // ---- layer 2 GEMM: z2 = [h1,h2] @ W2  (16 x 512 x 1024) ----
#pragma unroll
        for (int q = 0; q < 8; ++q) {
#pragma unroll
            for (int r = 0; r < 8; ++r) acc[q][r] = 0.0f;
        }
        for (int kt = 0; kt < 16; ++kt) {
            v16bf a = *(const v16bf*)&A2[kt * 512 + lane * 16];
            v16bf bfr[8];
#pragma unroll
            for (int q = 0; q < 8; ++q)
                bfr[q] = *(const v16bf*)&W2f[woff + (kt * 64 + (wv + 8 * q)) * 512 + lane * 16];
            __builtin_amdgcn_sched_barrier(0);
#pragma unroll
            for (int q = 0; q < 8; ++q)
                acc[q] = __builtin_amdgcn_wmma_f32_16x16x32_bf16(
                    false, a, false, bfr[q], (short)0, acc[q], false, false);
        }
        __syncthreads();   // all reads of A2 done before h2 rewrite

        // ---- layer 2 cell update; write h2 (bf16) into A2 K=256..511 ----
#pragma unroll
        for (int blk = 0; blk < 2; ++blk) {
            int n = (wv + 8 * blk) * 16 + lnN;
#pragma unroll
            for (int r = 0; r < 8; ++r) {
                float iv = acc[blk][r]     + bias2[blk];
                float jv = acc[blk + 2][r] + bias2[blk + 2];
                float fv = acc[blk + 4][r] + bias2[blk + 4];
                float ov = acc[blk + 6][r] + bias2[blk + 6];
                float nc = c2[blk][r] * sigm(fv + 1.0f) + sigm(iv) * tanh_fast(jv);
                float nh = tanh_fast(nc) * sigm(ov);
                c2[blk][r] = nc;
                int m = r + 8 * lh;
                st_afrag(A2, 256 + n, m, f2bf(nh));
            }
        }
        __syncthreads();
    }

    // ---- dense head: logits = h2 @ Wd + bd  (16 x 256 x 80) ----
    // h2 fragments already live in A2 k-tiles 8..15. 80 = 5 N-tiles.
    if (wv < 5) {
        v8f d;
#pragma unroll
        for (int r = 0; r < 8; ++r) d[r] = 0.0f;
        for (int kt = 0; kt < 8; ++kt) {
            v16bf a  = *(const v16bf*)&A2[(8 + kt) * 512 + lane * 16];
            v16bf bf = *(const v16bf*)&Wdf[(kt * 5 + wv) * 512 + lane * 16];
            d = __builtin_amdgcn_wmma_f32_16x16x32_bf16(
                false, a, false, bf, (short)0, d, false, false);
        }
        int n = wv * 16 + lnN;             // 0..79
        float bias = bd[n];
#pragma unroll
        for (int r = 0; r < 8; ++r) {
            int m = r + 8 * lh;
            out[(bid * 16 + m) * NUM_CLASSES + n] = d[r] + bias;
        }
    }
}

// ---------- launch ----------

extern "C" void kernel_launch(void* const* d_in, const int* in_sizes, int n_in,
                              void* d_out, int out_size, void* d_ws, size_t ws_size,
                              hipStream_t stream) {
    const int*   feats = (const int*)  d_in[0];
    const float* emb   = (const float*)d_in[1];
    const float* W1    = (const float*)d_in[2];
    const float* b1    = (const float*)d_in[3];
    const float* W2    = (const float*)d_in[4];
    const float* b2    = (const float*)d_in[5];
    const float* Wd    = (const float*)d_in[6];
    const float* bd    = (const float*)d_in[7];

    unsigned short* W1f = (unsigned short*)d_ws;              //  9*64*512
    unsigned short* W2f = W1f + 9 * 64 * 512;                 // 16*64*512
    unsigned short* Wdf = W2f + 16 * 64 * 512;                //  8*5*512

    {
        int total = 9 * 64 * 512;
        conv_w_kernel<<<(total + 255) / 256, 256, 0, stream>>>(W1, W1f, 9, 64, 4 * HID, EMB + HID);
    }
    {
        int total = 16 * 64 * 512;
        conv_w_kernel<<<(total + 255) / 256, 256, 0, stream>>>(W2, W2f, 16, 64, 4 * HID, 2 * HID);
    }
    {
        int total = 8 * 5 * 512;
        conv_w_kernel<<<(total + 255) / 256, 256, 0, stream>>>(Wd, Wdf, 8, 5, NUM_CLASSES, HID);
    }

    lstm_seq_kernel<<<BATCH / 16, 256, 0, stream>>>(
        feats, emb, W1f, b1, W2f, b2, Wdf, bd, (float*)d_out);
}
